// SpatialAttentionConv_13099650253566
// MI455X (gfx1250) — matrix operand order repaired and checked by reading, compiled, not run on hardware
//
#include <hip/hip_runtime.h>
#include <hip/hip_bf16.h>

typedef __attribute__((ext_vector_type(16))) _Float16 v16h;
typedef __attribute__((ext_vector_type(8)))  _Float16 v8h;
typedef __attribute__((ext_vector_type(8)))  float    v8f;

union H16 { v16h v; v8h h[2]; };
__device__ __forceinline__ v16h mk16(v8h lo, v8h hi) { H16 u; u.h[0] = lo; u.h[1] = hi; return u.v; }

// ---------------------------------------------------------------------------
// Prep kernels
// ---------------------------------------------------------------------------

// x (N x 128 fp32) -> f16 into columns [0,128) of A2 (N x 256 f16 panel)
__global__ void k_convert_x(const float* __restrict__ x, _Float16* __restrict__ A2, int N) {
  long t = (long)blockIdx.x * blockDim.x + threadIdx.x;
  long total = (long)N * 128;
  if (t >= total) return;
  long row = t >> 7; int c = (int)(t & 127);
  A2[row * 256 + c] = (_Float16)x[t];
}

// in_proj_w (384 x 128 fp32, [n][k]) -> f16 B for the QKV GEMM (same layout)
__global__ void k_cast_bqkv(const float* __restrict__ w, _Float16* __restrict__ Bqkv) {
  int t = blockIdx.x * blockDim.x + threadIdx.x;
  if (t < 384 * 128) Bqkv[t] = (_Float16)w[t];
}

// Fold output chain:  Bout[n][k<128] = W0[n][k];  Bout[n][128+k] = (W1 @ Wout)[n][k]
// bfold[n] = (W1 @ out_proj_b)[n] + conv_b[n]
// conv_w flat: [n][m][0][t] -> n*256 + m*2 + t   (t=0 -> W0, t=1 -> W1)
__global__ void k_fold_weights(const float* __restrict__ out_proj_w,
                               const float* __restrict__ out_proj_b,
                               const float* __restrict__ conv_w,
                               const float* __restrict__ conv_b,
                               _Float16* __restrict__ Bout,
                               float* __restrict__ bfold) {
  int n = threadIdx.x;  // one block, 128 threads
  for (int k = 0; k < 128; ++k)
    Bout[n * 256 + k] = (_Float16)conv_w[n * 256 + k * 2 + 0];
  for (int k = 0; k < 128; ++k) {
    float acc = 0.f;
    for (int m = 0; m < 128; ++m)
      acc += conv_w[n * 256 + m * 2 + 1] * out_proj_w[m * 128 + k];
    Bout[n * 256 + 128 + k] = (_Float16)acc;
  }
  float bacc = conv_b[n];
  for (int m = 0; m < 128; ++m)
    bacc += conv_w[n * 256 + m * 2 + 1] * out_proj_b[m];
  bfold[n] = bacc;
}

// ---------------------------------------------------------------------------
// WMMA GEMM:  C[M x Nn] = A[M x K] @ Bt[Nn x K]^T + bias
//   A: f16 row-major with leading dim lda (panel)
//   Bt: f16, Bt[n][k] contiguous in k (ldb == K)
//   block = 256 threads = 8 waves; block tile 128x128; wave tile 32x64
// ---------------------------------------------------------------------------
template <int K, bool OUT_F32>
__global__ __launch_bounds__(256) void k_gemm_wmma(
    const _Float16* __restrict__ A, int lda, int M,
    const _Float16* __restrict__ Bt,
    const float* __restrict__ bias,
    void* __restrict__ Cout, int ldc) {
  constexpr int KC = 64;   // k-chunk staged in LDS
  constexpr int KP = 88;   // LDS row pitch in halves: 176B, 16B aligned, conflict-free
  __shared__ _Float16 sA[128 * KP];
  __shared__ _Float16 sB[128 * KP];

  const int m0 = blockIdx.x * 128;
  const int n0 = blockIdx.y * 128;
  const int lane = threadIdx.x & 31;
  const int wave = threadIdx.x >> 5;
  const int wm = wave & 3;                 // 4 waves along M -> 32 rows each
  const int wn = wave >> 2;                // 2 waves along N -> 64 cols each
  const int lr = lane & 15;
  const int koffA = (lane >> 4) * 8;       // A: hi lanes hold K+8..15 / K+24..31
  const int koffB = (lane >> 4) * 8;       // B: hi lanes hold K+8..15 / K+24..31

  v8f acc[2][4];
#pragma unroll
  for (int i = 0; i < 2; ++i)
#pragma unroll
    for (int j = 0; j < 4; ++j)
#pragma unroll
      for (int r = 0; r < 8; ++r) acc[i][j][r] = 0.f;

  for (int kc = 0; kc < K; kc += KC) {
    // stage 128 x KC tiles of A (bounds-checked on M) and B
    for (int t = threadIdx.x; t < 128 * (KC / 8); t += 256) {
      int r = t >> 3;
      int kk = (t & 7) * 8;
      v8h av;
#pragma unroll
      for (int e = 0; e < 8; ++e) av[e] = (_Float16)0.f;
      int gr = m0 + r;
      if (gr < M) av = *(const v8h*)(A + (long)gr * lda + kc + kk);
      *(v8h*)(sA + r * KP + kk) = av;
      v8h bv = *(const v8h*)(Bt + (long)(n0 + r) * K + kc + kk);
      *(v8h*)(sB + r * KP + kk) = bv;
    }
    __syncthreads();
#pragma unroll
    for (int ks = 0; ks < KC; ks += 32) {
      v16h a[2], b[4];
#pragma unroll
      for (int i = 0; i < 2; ++i) {
        const _Float16* p = sA + (wm * 32 + i * 16 + lr) * KP + ks + koffA;
        a[i] = mk16(*(const v8h*)p, *(const v8h*)(p + 16));
      }
#pragma unroll
      for (int j = 0; j < 4; ++j) {
        const _Float16* p = sB + (wn * 64 + j * 16 + lr) * KP + ks + koffB;
        b[j] = mk16(*(const v8h*)p, *(const v8h*)(p + 16));
      }
#pragma unroll
      for (int i = 0; i < 2; ++i)
#pragma unroll
        for (int j = 0; j < 4; ++j)
          acc[i][j] = __builtin_amdgcn_wmma_f32_16x16x32_f16(
              false, a[i], false, b[j], (short)0, acc[i][j], false, false);
    }
    __syncthreads();
  }

  // epilogue: C/D layout -> VGPR r holds row (r or 8+r), col = lane&15
  const int rhalf = (lane >> 4) * 8;
#pragma unroll
  for (int i = 0; i < 2; ++i) {
#pragma unroll
    for (int j = 0; j < 4; ++j) {
      int col = n0 + wn * 64 + j * 16 + (lane & 15);
      int row0 = m0 + wm * 32 + i * 16 + rhalf;
      float bv = bias[col];
#pragma unroll
      for (int r = 0; r < 8; ++r) {
        int row = row0 + r;
        if (row < M) {
          float val = acc[i][j][r] + bv;
          if constexpr (OUT_F32)
            ((float*)Cout)[(long)row * ldc + col] = val;
          else
            ((_Float16*)Cout)[(long)row * ldc + col] = (_Float16)val;
        }
      }
    }
  }
}

// ---------------------------------------------------------------------------
// Per-face attention: 1 block = 1 face, 4 waves = 4 heads, lane = channel (Dh=32)
// QKV: N x 384 f16 (q|k|v). Writes o (f16) into cols [128,256) of A2 panel.
// ---------------------------------------------------------------------------
__global__ __launch_bounds__(128) void k_attn(const _Float16* __restrict__ QKV,
                                              const int* __restrict__ nbr,  // N x 9
                                              _Float16* __restrict__ A2, int N) {
  const long i = blockIdx.x;
  const int h = threadIdx.x >> 5;
  const int d = threadIdx.x & 31;
  const int c = h * 32 + d;
  const float scale = 0.1767766953f;  // 1/sqrt(32)

  float q = (float)QKV[i * 384 + c] * scale;

  float s[8], vreg[8];
#pragma unroll
  for (int j = 0; j < 8; ++j) {
    int idx = nbr[i * 9 + 1 + j];
    bool valid = ((unsigned)idx < (unsigned)N);
    float kd = 0.f, vd = 0.f;
    if (valid) {
      const _Float16* kr = QKV + (long)idx * 384;
      kd = (float)kr[128 + c];
      vd = (float)kr[256 + c];
    }
    vreg[j] = vd;
    float p = q * kd;
#pragma unroll
    for (int off = 16; off > 0; off >>= 1) p += __shfl_xor(p, off, 32);
    s[j] = valid ? p : -__builtin_huge_valf();
  }

  float m = s[0];
#pragma unroll
  for (int j = 1; j < 8; ++j) m = fmaxf(m, s[j]);
  float den = 0.f, w[8];
#pragma unroll
  for (int j = 0; j < 8; ++j) { w[j] = __expf(s[j] - m); den += w[j]; }
  float inv = 1.f / den;
  float o = 0.f;
#pragma unroll
  for (int j = 0; j < 8; ++j) o += w[j] * vreg[j];
  o *= inv;

  A2[i * 256 + 128 + c] = (_Float16)o;
}

// ---------------------------------------------------------------------------
extern "C" void kernel_launch(void* const* d_in, const int* in_sizes, int n_in,
                              void* d_out, int out_size, void* d_ws, size_t ws_size,
                              hipStream_t stream) {
  const float* x          = (const float*)d_in[0];
  const int*   nbr        = (const int*)d_in[1];
  const float* in_proj_w  = (const float*)d_in[4];
  const float* in_proj_b  = (const float*)d_in[5];
  const float* out_proj_w = (const float*)d_in[6];
  const float* out_proj_b = (const float*)d_in[7];
  const float* conv_w     = (const float*)d_in[8];
  const float* conv_b     = (const float*)d_in[9];
  const int N = in_sizes[0] / 128;

  // workspace layout (256B aligned slabs)
  char* ws = (char*)d_ws;
  size_t off = 0;
  auto alloc = [&](size_t bytes) -> void* {
    void* p = ws + off;
    off = (off + bytes + 255) & ~(size_t)255;
    return p;
  };
  _Float16* A2    = (_Float16*)alloc((size_t)N * 256 * sizeof(_Float16)); // [x | o] panel
  _Float16* QKV   = (_Float16*)alloc((size_t)N * 384 * sizeof(_Float16));
  _Float16* Bqkv  = (_Float16*)alloc((size_t)384 * 128 * sizeof(_Float16));
  _Float16* Bout  = (_Float16*)alloc((size_t)128 * 256 * sizeof(_Float16));
  float*    bfold = (float*)alloc(128 * sizeof(float));

  {
    long total = (long)N * 128;
    int blks = (int)((total + 255) / 256);
    k_convert_x<<<dim3(blks), 256, 0, stream>>>(x, A2, N);
  }
  k_cast_bqkv<<<dim3((384 * 128 + 255) / 256), 256, 0, stream>>>(in_proj_w, Bqkv);
  k_fold_weights<<<1, 128, 0, stream>>>(out_proj_w, out_proj_b, conv_w, conv_b, Bout, bfold);

  // QKV = x @ Wqkv^T + b   (N x 384, K = 128)
  dim3 gq((N + 127) / 128, 3);
  k_gemm_wmma<128, false><<<gq, 256, 0, stream>>>(A2, 256, N, Bqkv, in_proj_b, (void*)QKV, 384);

  // per-face softmax attention -> o into A2 cols [128,256)
  k_attn<<<dim3(N), 128, 0, stream>>>(QKV, nbr, A2, N);

  // out = [x | o] @ [W0 | W1*Wout]^T + bfold   (N x 128, K = 256), fp32 output
  dim3 go((N + 127) / 128, 1);
  k_gemm_wmma<256, true><<<go, 256, 0, stream>>>(A2, 256, N, Bout, bfold, d_out, 128);
}
